// DEMOHashLayer_3083786518797
// MI455X (gfx1250) — matrix operand order repaired and checked by reading, not compiled, not run on hardware
//
#include <hip/hip_runtime.h>
#include <hip/hip_bf16.h>

// ---------------------------------------------------------------------------
// Problem constants (match reference)
// ---------------------------------------------------------------------------
#define NN       50000
#define IN_DIM   256
#define OUT_DIM  512
#define NUM_HASH 8
#define HASH_DIM 256
#define KDIM     512            // concat(base, x) along K
#define MTILES   (NN / 16)      // 3125 (exact)

typedef __attribute__((ext_vector_type(16))) __bf16 v16bf;
typedef __attribute__((ext_vector_type(8)))  float  v8f;

__device__ __forceinline__ unsigned short f2bf(float f) {
    // round-to-nearest-even f32 -> bf16
    unsigned int u = __float_as_uint(f);
    unsigned int r = u + 0x7FFFu + ((u >> 16) & 1u);
    return (unsigned short)(r >> 16);
}

// ---------------------------------------------------------------------------
// 1) degree accumulation
// ---------------------------------------------------------------------------
__global__ void deg_kernel(const int* __restrict__ dst, float* __restrict__ deg, int e_count) {
    int e = blockIdx.x * blockDim.x + threadIdx.x;
    if (e < e_count) atomicAdd(deg + dst[e], 1.0f);
}

// ---------------------------------------------------------------------------
// 2) edge scatter: neigh_sum[dst] += x[src]   (64 threads/edge, float4 each)
// ---------------------------------------------------------------------------
__global__ void scatter_kernel(const float* __restrict__ x,
                               const int* __restrict__ src,
                               const int* __restrict__ dst,
                               float* __restrict__ nsum, int e_count) {
    int idx = blockIdx.x * blockDim.x + threadIdx.x;
    if (idx >= e_count * 64) return;
    int e = idx >> 6;
    int c = (idx & 63) << 2;           // dim offset 0,4,...,252
    int s = src[e];
    int d = dst[e];
    const float4 v = *(const float4*)(x + (size_t)s * IN_DIM + c);
    float* p = nsum + (size_t)d * IN_DIM + c;
    atomicAdd(p + 0, v.x);
    atomicAdd(p + 1, v.y);
    atomicAdd(p + 2, v.z);
    atomicAdd(p + 3, v.w);
}

// ---------------------------------------------------------------------------
// 3) build bf16 activation matrix Acat[N][512] = [ base | x ]
// ---------------------------------------------------------------------------
__global__ void build_acat(const float* __restrict__ x,
                           const float* __restrict__ nsum,
                           const float* __restrict__ deg,
                           unsigned short* __restrict__ Acat) {
    int n = blockIdx.x;
    int k = threadIdx.x;               // 0..255
    float dg = deg[n];
    float xv = x[(size_t)n * IN_DIM + k];
    float bv = (dg > 0.0f) ? (nsum[(size_t)n * IN_DIM + k] / dg) : xv;
    Acat[(size_t)n * KDIM + k]           = f2bf(bv);
    Acat[(size_t)n * KDIM + IN_DIM + k]  = f2bf(xv);
}

// ---------------------------------------------------------------------------
// 4) extract (col, sign) from dense signed one-hot hash matrices
// ---------------------------------------------------------------------------
__global__ void hash_extract(const float* __restrict__ hm,
                             int* __restrict__ col, float* __restrict__ sgn) {
    int t = blockIdx.x * blockDim.x + threadIdx.x;   // 0..2047  (h*256 + i)
    const float* row = hm + (size_t)t * HASH_DIM;
    int   c = 0;
    float s = 0.0f;
    for (int k = 0; k < HASH_DIM; ++k) {
        float v = row[k];
        if (v != 0.0f) { c = k; s = v; }
    }
    col[t] = c;
    sgn[t] = s;
}

// ---------------------------------------------------------------------------
// 5) build transposed fused weight  WT[j][r] (bf16), r<256: H@W_post, else W_self
// ---------------------------------------------------------------------------
__global__ void build_wt(const float* __restrict__ Wpost,
                         const float* __restrict__ Wself,
                         const int* __restrict__ col,
                         const float* __restrict__ sgn,
                         unsigned short* __restrict__ WT) {
    int j = blockIdx.x;                // output column 0..511
    int r = threadIdx.x;               // K row 0..511
    float acc;
    if (r < IN_DIM) {
        acc = 0.0f;
        #pragma unroll
        for (int h = 0; h < NUM_HASH; ++h) {
            int   c = col[h * IN_DIM + r];
            float s = sgn[h * IN_DIM + r];
            acc += s * Wpost[(size_t)(h * HASH_DIM + c) * OUT_DIM + j];
        }
    } else {
        acc = Wself[(size_t)(r - IN_DIM) * OUT_DIM + j];
    }
    WT[(size_t)j * KDIM + r] = f2bf(acc);
}

// ---------------------------------------------------------------------------
// 6) WMMA GEMM: out = elu(Acat @ Wcat + bias)
//    Block = 4 waves, one 64-col B panel staged in LDS (64KB), each wave
//    computes one 16-row M tile x four 16-col N tiles. K=512 unrolled ->
//    64 v_wmma_f32_16x16x32_bf16 per wave fed from ds_load_b128.
// ---------------------------------------------------------------------------
__global__ __launch_bounds__(128)
void gemm_wmma(const unsigned short* __restrict__ Acat,
               const unsigned short* __restrict__ WT,
               const float* __restrict__ bias,
               float* __restrict__ out) {
    __shared__ unsigned short Bs[64 * KDIM];     // 64KB B panel for this ngroup

    const int ngroup = blockIdx.y;               // 0..7, 64 output cols each

    // cooperative stage: WT[ngroup*64 .. +63][0..511]  (4096 x uint4)
    {
        const uint4* gsrc = (const uint4*)(WT + (size_t)ngroup * 64 * KDIM);
        uint4*       ldst = (uint4*)Bs;
        #pragma unroll 4
        for (int i = threadIdx.x; i < 4096; i += 128) ldst[i] = gsrc[i];
    }
    __syncthreads();

    const int wave  = threadIdx.x >> 5;
    const int lane  = threadIdx.x & 31;
    const int mtile = blockIdx.x * 4 + wave;
    if (mtile >= MTILES) return;                 // whole-wave guard, after barrier
    const int lrow  = lane & 15;
    const int lhalf = lane >> 4;

    // A fragment source: row (mtile*16 + lrow), K pattern per ISA 16-bit A layout
    const unsigned short* aRow = Acat + (size_t)(mtile * 16 + lrow) * KDIM + lhalf * 8;

    v8f acc[4];
    #pragma unroll
    for (int t = 0; t < 4; ++t) acc[t] = (v8f){};

    #pragma unroll
    for (int kb = 0; kb < KDIM; kb += 32) {
        union { v16bf v; uint4 u[2]; } a;
        a.u[0] = *(const uint4*)(aRow + kb);         // K = kb + lhalf*8 .. +7
        a.u[1] = *(const uint4*)(aRow + kb + 16);    // K = kb + 16 + lhalf*8 .. +7
        #pragma unroll
        for (int t = 0; t < 4; ++t) {
            const unsigned short* bCol =
                Bs + (t * 16 + lrow) * KDIM + kb + lhalf * 8;
            union { v16bf v; uint4 u[2]; } b;
            b.u[0] = *(const uint4*)(bCol);          // K = kb + lhalf*8 .. +7
            b.u[1] = *(const uint4*)(bCol + 16);     // K = kb + 16 + lhalf*8 .. +7
            acc[t] = __builtin_amdgcn_wmma_f32_16x16x32_bf16(
                         false, a.v, false, b.v, (short)0, acc[t], false, false);
        }
    }

    // Epilogue: bias + ELU, C layout: VGPR g -> M = g + lhalf*8, lane lrow -> N
    #pragma unroll
    for (int t = 0; t < 4; ++t) {
        int j = ngroup * 64 + t * 16 + lrow;
        float bj = bias[j];
        #pragma unroll
        for (int g = 0; g < 8; ++g) {
            int m = mtile * 16 + lhalf * 8 + g;
            float v = acc[t][g] + bj;
            out[(size_t)m * OUT_DIM + j] = (v > 0.0f) ? v : expm1f(v);
        }
    }
}

// ---------------------------------------------------------------------------
// launch
// ---------------------------------------------------------------------------
extern "C" void kernel_launch(void* const* d_in, const int* in_sizes, int n_in,
                              void* d_out, int out_size, void* d_ws, size_t ws_size,
                              hipStream_t stream) {
    const float* x         = (const float*)d_in[0];
    const float* W_post    = (const float*)d_in[1];
    const float* W_self    = (const float*)d_in[2];
    const float* bias      = (const float*)d_in[3];
    const float* hash_mats = (const float*)d_in[4];
    const int*   esrc      = (const int*)d_in[5];
    const int*   edst      = (const int*)d_in[6];
    float*       out       = (float*)d_out;

    // workspace layout (256B-aligned regions)
    char* ws = (char*)d_ws;
    float*          deg  = (float*)(ws);                          //    200,192 B
    float*          nsum = (float*)(ws + 200192);                 // 51,200,000 B
    unsigned short* Acat = (unsigned short*)(ws + 51400192);      // 51,200,000 B
    int*            col  = (int*)(ws + 102600192);                //      8,192 B
    float*          sgn  = (float*)(ws + 102608384);              //      8,192 B
    unsigned short* WT   = (unsigned short*)(ws + 102616576);     //    524,288 B
    // total ~103.1 MB

    const int e_count = in_sizes[5];

    // zero deg + neigh_sum (atomics accumulate into them)
    hipMemsetAsync(ws, 0, 200192 + (size_t)NN * IN_DIM * sizeof(float), stream);

    deg_kernel<<<(e_count + 255) / 256, 256, 0, stream>>>(edst, deg, e_count);

    {
        long long total = (long long)e_count * 64;
        int blocks = (int)((total + 255) / 256);
        scatter_kernel<<<blocks, 256, 0, stream>>>(x, esrc, edst, nsum, e_count);
    }

    build_acat<<<NN, IN_DIM, 0, stream>>>(x, nsum, deg, Acat);

    hash_extract<<<NUM_HASH, IN_DIM, 0, stream>>>(hash_mats, col, sgn);

    build_wt<<<OUT_DIM, KDIM, 0, stream>>>(W_post, W_self, col, sgn, WT);

    {
        dim3 grid((MTILES + 3) / 4, OUT_DIM / 64);   // (782, 8)
        gemm_wmma<<<grid, 128, 0, stream>>>(Acat, WT, bias, out);
    }
}
